// SyncedDiffLogicRecognizer_20701742367200
// MI455X (gfx1250) — compile-verified, matrix operand order checked by reading, not run on hardware
//
#include <hip/hip_runtime.h>
#include <hip/hip_bf16.h>

// ---------------- problem constants ----------------
#define Bb   128
#define Ss   512
#define Ee   1024
#define Hh   1024
#define Ll   4
#define GFf  2
#define Vv   128

// coefficient table layout in d_ws (rows of 4 floats):
//   [0,4096)      n layers  (L*H)
//   [4096,8192)   k layers
//   [8192,12288)  m layers
//   [12288,14336) m_fin     (H*GF)
#define COEF_N    0
#define COEF_K    (Ll*Hh)
#define COEF_M    (2*Ll*Hh)
#define COEF_FIN  (3*Ll*Hh)
#define COEF_ROWS (3*Ll*Hh + Hh*GFf)   // 14336

typedef __attribute__((ext_vector_type(16))) _Float16 v16h;
typedef __attribute__((ext_vector_type(8)))  float    v8f;
typedef __attribute__((__vector_size__(16))) int      vec4i;

// CDNA5 async-load-to-LDS availability probe
#if __has_builtin(__builtin_amdgcn_global_load_async_to_lds_b128) && \
    __has_builtin(__builtin_amdgcn_s_wait_asynccnt)
#define USE_ASYNC_LDS 1
#else
#define USE_ASYNC_LDS 0
#endif

// gate_g(a,b) = C[g][0] + C[g][1]*a + C[g][2]*b + C[g][3]*a*b
__constant__ float GATE_C[16][4] = {
    { 0.f,  0.f,  0.f,  0.f},   // 0
    { 0.f,  0.f,  0.f,  1.f},   // ab
    { 0.f,  1.f,  0.f, -1.f},   // a - ab
    { 0.f,  1.f,  0.f,  0.f},   // a
    { 0.f,  0.f,  1.f, -1.f},   // b - ab
    { 0.f,  0.f,  1.f,  0.f},   // b
    { 0.f,  1.f,  1.f, -2.f},   // a + b - 2ab
    { 0.f,  1.f,  1.f, -1.f},   // a + b - ab
    { 1.f, -1.f, -1.f,  1.f},   // 1 - (a+b-ab)
    { 1.f, -1.f, -1.f,  2.f},   // 1 - (a+b-2ab)
    { 1.f,  0.f, -1.f,  0.f},   // 1 - b
    { 1.f,  0.f, -1.f,  1.f},   // 1 - b + ab
    { 1.f, -1.f,  0.f,  0.f},   // 1 - a
    { 1.f, -1.f,  0.f,  1.f},   // 1 - a + ab
    { 1.f,  0.f,  0.f, -1.f},   // 1 - ab
    { 1.f,  0.f,  0.f,  0.f},   // 1
};

// ---------------------------------------------------------------------------
// Kernel 1: softmax(w[row,16]) @ GATE_C[16,4]  via  v_wmma_f32_16x16x32_f16.
// One wave per 16-row tile (verified emits v_wmma in round 1).
// ---------------------------------------------------------------------------
__global__ __launch_bounds__(256) void coef_wmma_kernel(
    const float* __restrict__ n_w, const float* __restrict__ k_w,
    const float* __restrict__ m_w, const float* __restrict__ fin_w,
    float* __restrict__ coef)
{
    const int wave = (blockIdx.x * blockDim.x + threadIdx.x) >> 5;
    const int lane = threadIdx.x & 31;
    const int r    = lane & 15;     // row within tile / output column N
    const int half = lane >> 4;     // which 8 of the 16 logits

    const int row0 = wave * 16;
    const int row  = row0 + r;

    const float* wsrc;
    if      (row < COEF_K)   wsrc = n_w   + (size_t)(row)            * 16;
    else if (row < COEF_M)   wsrc = k_w   + (size_t)(row - COEF_K)   * 16;
    else if (row < COEF_FIN) wsrc = m_w   + (size_t)(row - COEF_M)   * 16;
    else                     wsrc = fin_w + (size_t)(row - COEF_FIN) * 16;

    float wv[8];
#pragma unroll
    for (int j = 0; j < 8; ++j) wv[j] = wsrc[half * 8 + j];

    float mx = wv[0];
#pragma unroll
    for (int j = 1; j < 8; ++j) mx = fmaxf(mx, wv[j]);
    mx = fmaxf(mx, __shfl_xor(mx, 16, 32));

    float ex[8], sum = 0.f;
#pragma unroll
    for (int j = 0; j < 8; ++j) { ex[j] = __expf(wv[j] - mx); sum += ex[j]; }
    sum += __shfl_xor(sum, 16, 32);
    const float inv = 1.f / sum;

    v16h A;
#pragma unroll
    for (int j = 0; j < 8; ++j) A[j] = (_Float16)(ex[j] * inv);
#pragma unroll
    for (int j = 8; j < 16; ++j) A[j] = (_Float16)0.f;

    v16h Bm;
#pragma unroll
    for (int j = 0; j < 8; ++j) {
        const int K = half * 8 + j;
        Bm[j] = (r < 4) ? (_Float16)GATE_C[K][r] : (_Float16)0.f;
    }
#pragma unroll
    for (int j = 8; j < 16; ++j) Bm[j] = (_Float16)0.f;

    v8f Cacc = {};
    v8f D = __builtin_amdgcn_wmma_f32_16x16x32_f16(
        false, A, false, Bm, (short)0, Cacc, false, false);

    if (r < 4) {
#pragma unroll
        for (int rr = 0; rr < 8; ++rr) {
            const int orow = row0 + half * 8 + rr;
            coef[(size_t)orow * 4 + r] = D[rr];
        }
    }
}

// ---------------------------------------------------------------------------
// Kernel 1b: semb = sigmoid(emb_table)  (128x1024 = 512 KB, stays in L2)
// ---------------------------------------------------------------------------
__global__ __launch_bounds__(256) void semb_kernel(
    const float* __restrict__ emb, float* __restrict__ semb)
{
    const int i = blockIdx.x * blockDim.x + threadIdx.x;
    if (i < Vv * Ee) semb[i] = 1.f / (1.f + __expf(-emb[i]));
}

// ---------------------------------------------------------------------------
// Kernel 2: per-batch-element recurrence. One 1024-thread block per batch
// element. Per-thread layer (idx,coef) tuples are TIME-INVARIANT -> preloaded
// into registers; steady-state stage = 2 LDS gathers + 3 FMA + 1 LDS store.
// Next token's embedding row is async-prefetched into double-buffered LDS.
// ---------------------------------------------------------------------------
#if USE_ASYNC_LDS
__device__ __forceinline__ void async_copy16(const float* g, float* l) {
    __builtin_amdgcn_global_load_async_to_lds_b128(
        (__attribute__((address_space(1))) vec4i*)g,
        (__attribute__((address_space(3))) vec4i*)l,
        0, 0);
}
#endif

#define STAGE(SRC, DST, IA, IB, C)                                           \
    do {                                                                     \
        const float a_ = (SRC)[IA];                                          \
        const float b_ = (SRC)[IB];                                          \
        (DST)[tid] = fmaf((C).w, a_ * b_,                                    \
                     fmaf((C).z, b_, fmaf((C).y, a_, (C).x)));               \
        __syncthreads();                                                     \
    } while (0)

__global__ __launch_bounds__(1024) void recur_kernel(
    const int*   __restrict__ x,          // [B,S]
    const int*   __restrict__ last_index, // [B]
    const float* __restrict__ semb,       // [V,E] sigmoid(emb_table)
    const float4* __restrict__ coef,      // [COEF_ROWS]
    const float* __restrict__ rec_w,      // [H]
    const float* __restrict__ rec_b,      // [1]
    const int*   __restrict__ n_idx,      // [L,2,H]
    const int*   __restrict__ k_idx0,     // [2,H]
    const int*   __restrict__ k_idx_rest, // [L-1,2,H]
    const int*   __restrict__ m_idx0,     // [2,H]
    const int*   __restrict__ m_idx_rest, // [L-1,2,H]
    const int*   __restrict__ m_idx_fin,  // [2,H*GF]
    float*       __restrict__ out)        // [B]
{
    __shared__ float comb[2 * Hh];   // [0,H) = n, [H,2H) = h
    __shared__ float tmp [Hh];
    __shared__ float mbuf[Hh];
    __shared__ float red [32];
#if USE_ASYNC_LDS
    __shared__ float ebuf[2][Ee];
#endif

    const int b   = blockIdx.x;
    const int tid = threadIdx.x;
    const int li  = last_index[b];

    // ---- preload time-invariant per-thread layer constants into registers
    int   nia[Ll], nib[Ll];  float4 nc[Ll];
#pragma unroll
    for (int i = 0; i < Ll; ++i) {
        nia[i] = n_idx[i * 2 * Hh + tid];
        nib[i] = n_idx[i * 2 * Hh + Hh + tid];
        nc[i]  = coef[COEF_N + i * Hh + tid];
    }
    int   kia[Ll], kib[Ll];  float4 kc[Ll];
    kia[0] = k_idx0[tid];  kib[0] = k_idx0[Hh + tid];  kc[0] = coef[COEF_K + tid];
#pragma unroll
    for (int i = 1; i < Ll; ++i) {
        kia[i] = k_idx_rest[(i - 1) * 2 * Hh + tid];
        kib[i] = k_idx_rest[(i - 1) * 2 * Hh + Hh + tid];
        kc[i]  = coef[COEF_K + i * Hh + tid];
    }

    comb[Hh + tid] = 0.f;            // h0 = 0
    if (li == 0) comb[tid] = 0.f;    // m_block input = zeros

#if USE_ASYNC_LDS
    if (li > 0 && tid < 256) {       // prefetch row for t = 0
        const int tok0 = x[b * Ss];
        async_copy16(semb + (size_t)tok0 * Ee + tid * 4, &ebuf[0][tid * 4]);
    }
#endif
    __syncthreads();

    bool do_m = (li == 0);

    for (int t = 0; t < li; ++t) {
#if USE_ASYNC_LDS
        if (t + 1 < li) {            // prefetch row for t+1, overlap 8 stages
            const int tnx = x[b * Ss + t + 1];
            if (tid < 256)
                async_copy16(semb + (size_t)tnx * Ee + tid * 4,
                             &ebuf[(t + 1) & 1][tid * 4]);
            __builtin_amdgcn_s_wait_asynccnt(1);   // row for t is complete
        } else {
            __builtin_amdgcn_s_wait_asynccnt(0);
        }
        __syncthreads();
        const float* e = ebuf[t & 1];
#else
        const int tok = x[b * Ss + t];
        const float* e = semb + (size_t)tok * Ee;  // L2 gathers in n0
#endif

        // n layers: e -> tmp -> mbuf -> tmp -> comb_lo
        STAGE(e,    tmp,  nia[0], nib[0], nc[0]);
        STAGE(tmp,  mbuf, nia[1], nib[1], nc[1]);
        STAGE(mbuf, tmp,  nia[2], nib[2], nc[2]);
        STAGE(tmp,  comb, nia[3], nib[3], nc[3]);

        if (t + 1 == li) { do_m = true; break; }   // k of last step is unused

        // k layers: comb(2H) -> tmp -> mbuf -> tmp -> comb_hi  (h = k)
        STAGE(comb, tmp,       kia[0], kib[0], kc[0]);
        STAGE(tmp,  mbuf,      kia[1], kib[1], kc[1]);
        STAGE(mbuf, tmp,       kia[2], kib[2], kc[2]);
        STAGE(tmp,  comb + Hh, kia[3], kib[3], kc[3]);
    }

    if (do_m) {
        // m_block on comb = [n_t, h_t]: 4 layers + fin + GF-sum + logit
        {
            const int ia = m_idx0[tid], ib = m_idx0[Hh + tid];
            const float4 c = coef[COEF_M + tid];
            STAGE(comb, tmp, ia, ib, c);
        }
#pragma unroll
        for (int i = 1; i < Ll; ++i) {
            const int ia = m_idx_rest[(i - 1) * 2 * Hh + tid];
            const int ib = m_idx_rest[(i - 1) * 2 * Hh + Hh + tid];
            const float4 c = coef[COEF_M + i * Hh + tid];
            if (i & 1) { STAGE(tmp, mbuf, ia, ib, c); }
            else       { STAGE(mbuf, tmp, ia, ib, c); }
        }
        // after m1,m2,m3 the result sits in mbuf

        float acc = 0.f;
#pragma unroll
        for (int gf = 0; gf < GFf; ++gf) {
            const int o  = tid * GFf + gf;
            const int ia = m_idx_fin[o];
            const int ib = m_idx_fin[Hh * GFf + o];
            const float4 c = coef[COEF_FIN + o];
            const float a = mbuf[ia];
            const float bb = mbuf[ib];
            acc += fmaf(c.w, a * bb, fmaf(c.z, bb, fmaf(c.y, a, c.x)));
        }
        float val = acc * rec_w[tid];

        // block reduce (1024 -> 1)
#pragma unroll
        for (int off = 16; off > 0; off >>= 1) val += __shfl_down(val, off, 32);
        if ((tid & 31) == 0) red[tid >> 5] = val;
        __syncthreads();
        if (tid < 32) {
            float v = red[tid];
#pragma unroll
            for (int off = 16; off > 0; off >>= 1) v += __shfl_down(v, off, 32);
            if (tid == 0) out[b] = v + rec_b[0];
        }
    }
}

// ---------------------------------------------------------------------------
extern "C" void kernel_launch(void* const* d_in, const int* in_sizes, int n_in,
                              void* d_out, int out_size, void* d_ws, size_t ws_size,
                              hipStream_t stream) {
    const int*   x          = (const int*)  d_in[0];
    const int*   last_index = (const int*)  d_in[1];
    // d_in[2] positive_mask: unused by the reference computation
    const float* emb_table  = (const float*)d_in[3];
    const float* n_w        = (const float*)d_in[4];
    const float* k_w        = (const float*)d_in[5];
    const float* m_w        = (const float*)d_in[6];
    const float* m_fin_w    = (const float*)d_in[7];
    const float* rec_w      = (const float*)d_in[8];
    const float* rec_b      = (const float*)d_in[9];
    const int*   n_idx      = (const int*)  d_in[10];
    const int*   k_idx0     = (const int*)  d_in[11];
    const int*   k_idx_rest = (const int*)  d_in[12];
    const int*   m_idx0     = (const int*)  d_in[13];
    const int*   m_idx_rest = (const int*)  d_in[14];
    const int*   m_idx_fin  = (const int*)  d_in[15];

    float* coef = (float*)d_ws;                          // 229,376 B
    float* semb = (float*)((char*)d_ws + COEF_ROWS * 16); // +524,288 B

    coef_wmma_kernel<<<112, 256, 0, stream>>>(n_w, k_w, m_w, m_fin_w, coef);
    semb_kernel<<<(Vv * Ee) / 256, 256, 0, stream>>>(emb_table, semb);

    recur_kernel<<<Bb, 1024, 0, stream>>>(
        x, last_index, semb, (const float4*)coef, rec_w, rec_b,
        n_idx, k_idx0, k_idx_rest, m_idx0, m_idx_rest, m_idx_fin,
        (float*)d_out);
}
